// OptimizedDynamicSparseAttention_13932873908493
// MI455X (gfx1250) — matrix operand order repaired and checked by reading, not compiled, not run on hardware
//
#include <hip/hip_runtime.h>
#include <cstdint>
#include <cstddef>

// Problem constants (match reference)
#define B_    2
#define S_    1024
#define H_    2048
#define NH_   16
#define NKV_  8
#define HD_   128
#define QKVN  (NH_*HD_ + 2*NKV_*HD_)   // 4096
#define TOPK_ 409

// GEMM tiling
#define TM 64
#define TN 128
#define TK 32

typedef float        v8f   __attribute__((ext_vector_type(8)));
typedef __bf16       v16bf __attribute__((ext_vector_type(16)));
typedef unsigned int v8u   __attribute__((ext_vector_type(8)));
typedef unsigned int v4u   __attribute__((ext_vector_type(4)));
typedef unsigned int u32x4 __attribute__((ext_vector_type(4)));
typedef int          i32x8 __attribute__((ext_vector_type(8)));
typedef int          i32x4 __attribute__((ext_vector_type(4)));

// Tensor Data Mover availability (device pass only; host pass sees fallback)
#if defined(__gfx1250__) && __has_builtin(__builtin_amdgcn_tensor_load_to_lds)
#define USE_TDM 1
#else
#define USE_TDM 0
#endif

// ---------------------------------------------------------------------------
// WMMA fragment loaders (contiguous k order, same in A and B; wave32)
// A (16xK, row-major source): lane<16 -> M=lane, K = k0..k0+15
//                             lane>=16 -> M=lane-16, K = k0+16..k0+31
// ---------------------------------------------------------------------------
__device__ __forceinline__ v16bf load_fragA(const __bf16* base, int stride, int k0) {
    const int lane = threadIdx.x & 31;
    const int m    = lane & 15;
    const int kk   = k0 + ((lane >> 4) << 4);
    const unsigned int* p0 = reinterpret_cast<const unsigned int*>(base + (size_t)m * stride + kk);
    const unsigned int* p1 = reinterpret_cast<const unsigned int*>(base + (size_t)m * stride + kk + 8);
    v8u u;
    u[0] = p0[0]; u[1] = p0[1]; u[2] = p0[2]; u[3] = p0[3];
    u[4] = p1[0]; u[5] = p1[1]; u[6] = p1[2]; u[7] = p1[3];
    return __builtin_bit_cast(v16bf, u);
}

// B (Kx16): element (k,n) at base[n*stride + k]
// lane<16 -> N=lane, K = k0..k0+15 ; lane>=16 -> N=lane-16, K = k0+16..k0+31
__device__ __forceinline__ v16bf load_fragB(const __bf16* base, int stride, int k0) {
    const int lane = threadIdx.x & 31;
    const int n    = lane & 15;
    const int kk   = k0 + ((lane >> 4) << 4);
    const unsigned int* p = reinterpret_cast<const unsigned int*>(base + (size_t)n * stride + kk);
    v8u u;
#pragma unroll
    for (int i = 0; i < 8; ++i) u[i] = p[i];
    return __builtin_bit_cast(v16bf, u);
}

// ---------------------------------------------------------------------------
// wave32 reductions
// ---------------------------------------------------------------------------
__device__ __forceinline__ float wave_max(float v) {
#pragma unroll
    for (int off = 16; off > 0; off >>= 1) v = fmaxf(v, __shfl_xor(v, off, 32));
    return v;
}
__device__ __forceinline__ float wave_sum(float v) {
#pragma unroll
    for (int off = 16; off > 0; off >>= 1) v += __shfl_xor(v, off, 32);
    return v;
}
__device__ __forceinline__ int wave_isum(int v) {
#pragma unroll
    for (int off = 16; off > 0; off >>= 1) v += __shfl_xor(v, off, 32);
    return v;
}

// ---------------------------------------------------------------------------
// TDM: issue a 2D tile load (rows x cols bf16 elements) into LDS.
// D# built per ISA 08_async_tensor: group0 = {flags, lds_addr, gaddr_lo, gaddr_hi|type},
// group1 = {data_size, tensor dims/strides, tile dims}.
// ---------------------------------------------------------------------------
#if USE_TDM
__device__ __forceinline__ void tdm_load_tile(unsigned lds_addr, const __bf16* gptr,
                                              int cols, int rows, int tensor_rows,
                                              int row_stride_elems) {
    const unsigned long long ga = (unsigned long long)(uintptr_t)gptr;
    u32x4 g0;
    g0[0] = 1u;                                             // count=1, user descriptor
    g0[1] = lds_addr;                                       // LDS byte address
    g0[2] = (unsigned)(ga & 0xFFFFFFFFu);                   // global_addr[31:0]
    g0[3] = ((unsigned)(ga >> 32) & 0x01FFFFFFu) | (2u << 30); // global_addr[56:32], type=2
    i32x8 g1;
    g1[0] = (int)(1u << 16);                                // workgroup_mask=0, data_size=1 (2B)
    g1[1] = (int)(((unsigned)row_stride_elems & 0xFFFFu) << 16);          // tensor_dim0 lo16
    g1[2] = (int)((((unsigned)row_stride_elems >> 16) & 0xFFFFu)
                | (((unsigned)tensor_rows & 0xFFFFu) << 16));             // dim0 hi / dim1 lo
    g1[3] = (int)((((unsigned)tensor_rows >> 16) & 0xFFFFu)
                | (((unsigned)cols & 0xFFFFu) << 16));                    // dim1 hi / tile_dim0
    g1[4] = (int)(unsigned)rows;                            // tile_dim1 = rows, tile_dim2 = 0
    g1[5] = (int)(unsigned)row_stride_elems;                // tensor_dim0_stride lo32
    g1[6] = 0;
    g1[7] = 0;
    i32x4 z = {0, 0, 0, 0};
#if __clang_major__ >= 23
    i32x8 z8 = {0, 0, 0, 0, 0, 0, 0, 0};
    __builtin_amdgcn_tensor_load_to_lds(g0, g1, z, z, z8, 0);
#else
    __builtin_amdgcn_tensor_load_to_lds(g0, g1, z, z, 0);
#endif
}
#endif

// ---------------------------------------------------------------------------
// fp32 -> bf16 conversion
// ---------------------------------------------------------------------------
__global__ void cvt_f32_bf16(const float* __restrict__ src, __bf16* __restrict__ dst, size_t n) {
    size_t i = (size_t)blockIdx.x * blockDim.x + threadIdx.x;
    if (i < n) dst[i] = (__bf16)src[i];
}

// ---------------------------------------------------------------------------
// Generic WMMA GEMM: C[m,n] = sum_k A[m,k]*W[n,k] (+ bias[n]), C fp32
// Block = 256 thr (8 waves). Tile TM=64 x TN=128, K step 32.
// LDS tiles double-buffered and fed by TDM (tensor_load_to_lds) when available.
// ---------------------------------------------------------------------------
__global__ void __launch_bounds__(256) gemm_bias_bf16(
        const __bf16* __restrict__ A, const __bf16* __restrict__ W,
        const float* __restrict__ bias, float* __restrict__ Cout,
        int M, int N, int K)
{
    __shared__ __bf16 sA[2][TM * TK];   // 2 x 4 KB
    __shared__ __bf16 sB[2][TN * TK];   // 2 x 8 KB
    const int tid   = threadIdx.x;
    const int wave  = tid >> 5;
    const int lane  = tid & 31;
    const int mBase = blockIdx.y * TM;
    const int nBase = blockIdx.x * TN;
    const int m_off  = (wave >> 1) * 16;      // 0,16,32,48
    const int n_base = (wave & 1) * 64;       // 0 or 64

    v8f acc[4];
#pragma unroll
    for (int j = 0; j < 4; ++j) acc[j] = (v8f){0.f,0.f,0.f,0.f,0.f,0.f,0.f,0.f};

    const int steps = K / TK;

#if USE_TDM
    if (wave == 0) {
        tdm_load_tile((unsigned)(uintptr_t)(&sA[0][0]), A + (size_t)mBase * K, TK, TM, M, K);
        tdm_load_tile((unsigned)(uintptr_t)(&sB[0][0]), W + (size_t)nBase * K, TK, TN, N, K);
    }
#endif

#pragma unroll 2
    for (int it = 0; it < steps; ++it) {
        const int cur = it & 1;
#if USE_TDM
        if (wave == 0) {
            if (it + 1 < steps) {
                const int k1 = (it + 1) * TK;
                tdm_load_tile((unsigned)(uintptr_t)(&sA[cur ^ 1][0]),
                              A + (size_t)mBase * K + k1, TK, TM, M, K);
                tdm_load_tile((unsigned)(uintptr_t)(&sB[cur ^ 1][0]),
                              W + (size_t)nBase * K + k1, TK, TN, N, K);
                __builtin_amdgcn_s_wait_tensorcnt((short)2);
            } else {
                __builtin_amdgcn_s_wait_tensorcnt((short)0);
            }
        }
#else
        {   // cooperative fallback load of current K tile
            const int k0 = it * TK;
            {
                const int r = tid >> 2, c = (tid & 3) << 3;
                *reinterpret_cast<v4u*>(&sA[cur][0] + r * TK + c) =
                    *reinterpret_cast<const v4u*>(A + (size_t)(mBase + r) * K + k0 + c);
            }
#pragma unroll
            for (int j = 0; j < 2; ++j) {
                const int idx = tid * 2 + j;
                const int r = idx >> 2, c = (idx & 3) << 3;
                *reinterpret_cast<v4u*>(&sB[cur][0] + r * TK + c) =
                    *reinterpret_cast<const v4u*>(W + (size_t)(nBase + r) * K + k0 + c);
            }
            if (k0 + TK < K) {
                const int r = tid >> 2, c = (tid & 3) << 3;
                __builtin_prefetch(A + (size_t)(mBase + r) * K + k0 + TK + c, 0, 1);
                __builtin_prefetch(W + (size_t)(nBase + r) * K + k0 + TK + c, 0, 1);
            }
        }
#endif
        __syncthreads();
        v16bf a = load_fragA(&sA[cur][0] + m_off * TK, TK, 0);
#pragma unroll
        for (int j = 0; j < 4; ++j) {
            v16bf b = load_fragB(&sB[cur][0] + (n_base + j * 16) * TK, TK, 0);
            acc[j] = __builtin_amdgcn_wmma_f32_16x16x32_bf16(false, a, false, b, (short)0, acc[j], false, false);
        }
        __syncthreads();
    }

    const int m0 = mBase + m_off + ((lane >> 4) << 3);
#pragma unroll
    for (int j = 0; j < 4; ++j) {
        const int n0 = nBase + n_base + j * 16 + (lane & 15);
        const float bvj = bias ? bias[n0] : 0.0f;
#pragma unroll
        for (int rr = 0; rr < 8; ++rr)
            Cout[(size_t)(m0 + rr) * N + n0] = acc[j][rr] + bvj;
    }
}

// ---------------------------------------------------------------------------
// Bias add + RoPE + pack. Q,K -> (B, heads, S, HD). V -> TRANSPOSED (B,NKV,HD,S)
// so that the P@V B-fragment reads contiguous keys.
// ---------------------------------------------------------------------------
__global__ void rope_pack(const float* __restrict__ QKV,
                          const float* __restrict__ bq, const float* __restrict__ bk,
                          const float* __restrict__ bv, const int* __restrict__ pos,
                          __bf16* __restrict__ Q, __bf16* __restrict__ Kc, __bf16* __restrict__ Vt)
{
    size_t idx = (size_t)blockIdx.x * blockDim.x + threadIdx.x;
    const size_t total = (size_t)B_ * S_ * QKVN;
    if (idx >= total) return;
    const int f = (int)(idx % QKVN);
    const int m = (int)(idx / QKVN);
    const int b = m / S_, s = m % S_;
    const int p = pos[m];
    float x = QKV[idx];

    if (f < NH_ * HD_) {                        // Q
        const int h = f / HD_, d = f % HD_;
        x += bq[f];
        const int dp = (d < 64) ? d + 64 : d - 64;
        const float xp  = QKV[(size_t)m * QKVN + h * HD_ + dp] + bq[h * HD_ + dp];
        const float sgn = (d < 64) ? -1.f : 1.f;
        const int fi = d & 63;
        const float inv = __powf(1.0e6f, -(float)(2 * fi) * (1.0f / 128.0f));
        float sn, cs; __sincosf((float)p * inv, &sn, &cs);
        Q[(((size_t)(b * NH_ + h)) * S_ + s) * HD_ + d] = (__bf16)(x * cs + sgn * xp * sn);
    } else if (f < NH_ * HD_ + NKV_ * HD_) {    // K
        const int fk = f - NH_ * HD_;
        const int kvh = fk / HD_, d = fk % HD_;
        x += bk[fk];
        const int dp = (d < 64) ? d + 64 : d - 64;
        const float xp  = QKV[(size_t)m * QKVN + NH_ * HD_ + kvh * HD_ + dp] + bk[kvh * HD_ + dp];
        const float sgn = (d < 64) ? -1.f : 1.f;
        const int fi = d & 63;
        const float inv = __powf(1.0e6f, -(float)(2 * fi) * (1.0f / 128.0f));
        float sn, cs; __sincosf((float)p * inv, &sn, &cs);
        Kc[(((size_t)(b * NKV_ + kvh)) * S_ + s) * HD_ + d] = (__bf16)(x * cs + sgn * xp * sn);
    } else {                                    // V: bias + transpose to (b,kv,d,s)
        const int fv = f - NH_ * HD_ - NKV_ * HD_;
        const int kvh = fv / HD_, d = fv % HD_;
        Vt[((size_t)(b * NKV_ + kvh) * HD_ + d) * S_ + s] = (__bf16)(x + bv[fv]);
    }
}

// ---------------------------------------------------------------------------
// Attention: one block per (b, h, 16-query tile). 256 thr = 8 waves.
// K and V chunks staged into LDS by TDM (double-buffered, 128 keys per chunk);
// full 16x1024 score row-block in LDS -> top-k threshold -> softmax -> P@V.
// Dynamic LDS: 64KB scores + 32KB probs + 2x32KB stage buffers = 160KB.
// ---------------------------------------------------------------------------
#define KCHUNK 128
#define NCHUNKS (S_ / KCHUNK)   // 8

__global__ void __launch_bounds__(256) attn_kernel(
        const __bf16* __restrict__ Q, const __bf16* __restrict__ Kb,
        const __bf16* __restrict__ Vt, __bf16* __restrict__ O)
{
    extern __shared__ __align__(16) char smem[];
    float*  sc   = (float*)smem;                                      // [16][1024]   64KB
    __bf16* pb   = (__bf16*)(smem + (size_t)16 * S_ * sizeof(float)); // [16][1024]   32KB
    __bf16* stg0 = pb + (size_t)16 * S_;                              // [128][128]   32KB
    __bf16* stg1 = stg0 + (size_t)KCHUNK * HD_;                       // [128][128]   32KB

    const int qblk = blockIdx.x;
    const int h    = blockIdx.y;
    const int b    = blockIdx.z;
    const int kv   = h >> 1;   // GROUPS = 2
    const int tid  = threadIdx.x;
    const int wave = tid >> 5;
    const int lane = tid & 31;

    const __bf16* Qh  = Q  + (((size_t)(b * NH_  + h ) * S_) + qblk * 16) * HD_;
    const __bf16* Kh  = Kb + (  (size_t)(b * NKV_ + kv) * S_) * HD_;
    const __bf16* Vth = Vt + (  (size_t)(b * NKV_ + kv) * HD_) * S_;

    // Preload the 16x128 Q tile as 4 A-fragments (K chunks of 32)
    v16bf qf[4];
#pragma unroll
    for (int cix = 0; cix < 4; ++cix) qf[cix] = load_fragA(Qh, HD_, cix * 32);

    const float scale = 0.08838834764831845f;  // 1/sqrt(128)

    // ---- scores: 8 chunks of 128 keys; K chunk staged by TDM, double-buffered ----
#if USE_TDM
    if (wave == 0) {
        // K chunk is contiguous (row-major (S,HD)): 128x128 tile, stride HD
        tdm_load_tile((unsigned)(uintptr_t)stg0, Kh, HD_, KCHUNK, S_, HD_);
    }
#endif
#pragma unroll 2
    for (int chunk = 0; chunk < NCHUNKS; ++chunk) {
        __bf16* curK = (chunk & 1) ? stg1 : stg0;
#if USE_TDM
        if (wave == 0) {
            if (chunk + 1 < NCHUNKS) {
                __bf16* nxtK = (chunk & 1) ? stg0 : stg1;
                tdm_load_tile((unsigned)(uintptr_t)nxtK,
                              Kh + (size_t)(chunk + 1) * KCHUNK * HD_, HD_, KCHUNK, S_, HD_);
                __builtin_amdgcn_s_wait_tensorcnt((short)1);
            } else {
                __builtin_amdgcn_s_wait_tensorcnt((short)0);
            }
        }
#else
        {   // cooperative fallback: copy 128x128 contiguous chunk (32KB)
            const __bf16* src = Kh + (size_t)chunk * KCHUNK * HD_;
#pragma unroll
            for (int j = 0; j < 8; ++j) {
                const int e = (tid + j * 256) * 8;   // 8 bf16 per thread per step
                *reinterpret_cast<v4u*>(curK + e) = *reinterpret_cast<const v4u*>(src + e);
            }
        }
#endif
        __syncthreads();
        const int keyL = wave * 16;                 // wave's 16 keys within chunk
        v8f acc = {0.f,0.f,0.f,0.f,0.f,0.f,0.f,0.f};
#pragma unroll
        for (int cix = 0; cix < 4; ++cix) {
            v16bf bf = load_fragB(curK + (size_t)keyL * HD_, HD_, cix * 32);
            acc = __builtin_amdgcn_wmma_f32_16x16x32_bf16(false, qf[cix], false, bf, (short)0, acc, false, false);
        }
        const int n  = chunk * KCHUNK + keyL + (lane & 15);
        const int mr = (lane >> 4) << 3;
#pragma unroll
        for (int r = 0; r < 8; ++r) sc[(size_t)(mr + r) * S_ + n] = acc[r] * scale;
        __syncthreads();
    }

#if USE_TDM
    // Kick off the first V chunk; DMA overlaps the top-k/softmax VALU work.
    // Vt is (HD, S): 128-row x 128-col tile with row stride S.
    if (wave == 0) {
        tdm_load_tile((unsigned)(uintptr_t)stg0, Vth, KCHUNK, HD_, HD_, S_);
    }
#endif

    // ---- top-k threshold + masked softmax; each wave handles 2 rows ----
#pragma unroll 1
    for (int rr = 0; rr < 2; ++rr) {
        const int row = wave * 2 + rr;
        float vals[32];
        float vmax = -3.0e38f, vmin = 3.0e38f;
#pragma unroll
        for (int i = 0; i < 32; ++i) {
            float v = sc[(size_t)row * S_ + lane * 32 + i];
            vals[i] = v;
            vmax = fmaxf(vmax, v);
            vmin = fminf(vmin, v);
        }
        vmax = wave_max(vmax);
        vmin = -wave_max(-vmin);

        float lo = vmin, hi = vmax;
#pragma unroll 1
        for (int itb = 0; itb < 24; ++itb) {
            const float mid = 0.5f * (lo + hi);
            int cnt = 0;
#pragma unroll
            for (int i = 0; i < 32; ++i) cnt += (vals[i] >= mid) ? 1 : 0;
            cnt = wave_isum(cnt);
            if (cnt >= TOPK_) lo = mid; else hi = mid;
        }
        const float th = lo;

        float sum = 0.f;
#pragma unroll
        for (int i = 0; i < 32; ++i)
            sum += (vals[i] >= th) ? __expf(vals[i] - vmax) : 0.f;
        sum = wave_sum(sum);
        const float rinv = 1.0f / sum;
#pragma unroll
        for (int i = 0; i < 32; ++i) {
            const float pr = (vals[i] >= th) ? __expf(vals[i] - vmax) * rinv : 0.f;
            pb[(size_t)row * S_ + lane * 32 + i] = (__bf16)pr;
        }
    }
    __syncthreads();

    // ---- P @ V: wave owns d-columns [wave*16, wave*16+16). V chunk staged by TDM ----
    const int d0 = wave * 16;
    v8f oacc = {0.f,0.f,0.f,0.f,0.f,0.f,0.f,0.f};
#pragma unroll 2
    for (int chunk = 0; chunk < NCHUNKS; ++chunk) {
        __bf16* curV = (chunk & 1) ? stg1 : stg0;   // [HD][KCHUNK]: elem (d,kl) at d*KCHUNK+kl
#if USE_TDM
        if (wave == 0) {
            if (chunk + 1 < NCHUNKS) {
                __bf16* nxtV = (chunk & 1) ? stg0 : stg1;
                tdm_load_tile((unsigned)(uintptr_t)nxtV,
                              Vth + (size_t)(chunk + 1) * KCHUNK, KCHUNK, HD_, HD_, S_);
                __builtin_amdgcn_s_wait_tensorcnt((short)1);
            } else {
                __builtin_amdgcn_s_wait_tensorcnt((short)0);
            }
        }
#else
        {   // cooperative fallback: gather 128 rows of 128 keys (row stride S)
#pragma unroll
            for (int j = 0; j < 8; ++j) {
                const int e  = (tid + j * 256) * 8;
                const int dr = e / KCHUNK, kc = e % KCHUNK;
                *reinterpret_cast<v4u*>(curV + e) =
                    *reinterpret_cast<const v4u*>(Vth + (size_t)dr * S_ + chunk * KCHUNK + kc);
            }
        }
#endif
        __syncthreads();
#pragma unroll
        for (int sub = 0; sub < 4; ++sub) {
            v16bf a  = load_fragA(pb, S_, chunk * KCHUNK + sub * 32);
            v16bf bv = load_fragB(curV + (size_t)d0 * KCHUNK, KCHUNK, sub * 32);
            oacc = __builtin_amdgcn_wmma_f32_16x16x32_bf16(false, a, false, bv, (short)0, oacc, false, false);
        }
        __syncthreads();
    }

    const int n  = d0 + (lane & 15);
    const int mr = (lane >> 4) << 3;
#pragma unroll
    for (int r = 0; r < 8; ++r) {
        const int s = qblk * 16 + mr + r;
        O[((size_t)(b * S_ + s)) * (NH_ * HD_) + h * HD_ + n] = (__bf16)oacc[r];
    }
}

// ---------------------------------------------------------------------------
// Host-side launcher
// ---------------------------------------------------------------------------
extern "C" void kernel_launch(void* const* d_in, const int* in_sizes, int n_in,
                              void* d_out, int out_size, void* d_ws, size_t ws_size,
                              hipStream_t stream)
{
    (void)in_sizes; (void)n_in; (void)out_size; (void)ws_size;

    const float* hidden = (const float*)d_in[0];
    const float* wq = (const float*)d_in[1];
    const float* bq = (const float*)d_in[2];
    const float* wk = (const float*)d_in[3];
    const float* bk = (const float*)d_in[4];
    const float* wv = (const float*)d_in[5];
    const float* bv = (const float*)d_in[6];
    const float* wo = (const float*)d_in[7];
    const float* bo = (const float*)d_in[8];
    const int*  pos = (const int*)d_in[9];
    float* out = (float*)d_out;

    char* p = (char*)d_ws;
    auto carve = [&](size_t bytes) -> char* {
        char* r = p;
        p += (bytes + 255) & ~(size_t)255;
        return r;
    };
    __bf16* Xbf  = (__bf16*)carve((size_t)B_ * S_ * H_ * 2);        // 8 MB
    __bf16* Wqkv = (__bf16*)carve((size_t)QKVN * H_ * 2);           // 16 MB
    __bf16* Wobf = (__bf16*)carve((size_t)H_ * H_ * 2);             // 8 MB
    float*  QKVf = (float*) carve((size_t)B_ * S_ * QKVN * 4);      // 32 MB
    __bf16* Qbf  = (__bf16*)carve((size_t)B_ * NH_  * S_ * HD_ * 2);// 8 MB
    __bf16* Kbf  = (__bf16*)carve((size_t)B_ * NKV_ * S_ * HD_ * 2);// 4 MB
    __bf16* Vtbf = (__bf16*)carve((size_t)B_ * NKV_ * S_ * HD_ * 2);// 4 MB
    __bf16* Obf  = (__bf16*)QKVf;  // alias: QKVf fully consumed by rope_pack before attn writes

    const int TB = 256;
    // 1) fp32 -> bf16 conversions
    {
        size_t n = (size_t)B_ * S_ * H_;
        cvt_f32_bf16<<<(unsigned)((n + TB - 1) / TB), TB, 0, stream>>>(hidden, Xbf, n);
        n = (size_t)NH_ * HD_ * H_;
        cvt_f32_bf16<<<(unsigned)((n + TB - 1) / TB), TB, 0, stream>>>(wq, Wqkv, n);
        n = (size_t)NKV_ * HD_ * H_;
        cvt_f32_bf16<<<(unsigned)((n + TB - 1) / TB), TB, 0, stream>>>(wk, Wqkv + (size_t)NH_ * HD_ * H_, n);
        cvt_f32_bf16<<<(unsigned)((n + TB - 1) / TB), TB, 0, stream>>>(wv, Wqkv + (size_t)(NH_ + NKV_) * HD_ * H_, n);
        n = (size_t)H_ * H_;
        cvt_f32_bf16<<<(unsigned)((n + TB - 1) / TB), TB, 0, stream>>>(wo, Wobf, n);
    }
    // 2) fused QKV projection (bias deferred to rope_pack)
    {
        dim3 grid(QKVN / TN, (B_ * S_) / TM);
        gemm_bias_bf16<<<grid, TB, 0, stream>>>(Xbf, Wqkv, nullptr, QKVf, B_ * S_, QKVN, H_);
    }
    // 3) bias + RoPE + pack (V transposed)
    {
        size_t n = (size_t)B_ * S_ * QKVN;
        rope_pack<<<(unsigned)((n + TB - 1) / TB), TB, 0, stream>>>(QKVf, bq, bk, bv, pos, Qbf, Kbf, Vtbf);
    }
    // 4) sparse attention (scores + top-k + softmax + P@V fused per query tile)
    {
        dim3 grid(S_ / 16, NH_, B_);
        size_t shmem = (size_t)16 * S_ * sizeof(float)    // scores
                     + (size_t)16 * S_ * sizeof(__bf16)   // probs
                     + (size_t)2 * KCHUNK * HD_ * sizeof(__bf16); // 2 stage buffers
        attn_kernel<<<grid, TB, shmem, stream>>>(Qbf, Kbf, Vtbf, Obf);
    }
    // 5) output projection
    {
        dim3 grid(H_ / TN, (B_ * S_) / TM);
        gemm_bias_bf16<<<grid, TB, 0, stream>>>(Obf, Wobf, bo, out, B_ * S_, H_, H_);
    }
}
